// Self_Attention_5274219839684
// MI455X (gfx1250) — compile-verified, matrix operand order checked by reading, not run on hardware
//
#include <hip/hip_runtime.h>
#include <hip/hip_bf16.h>

// ---------------------------------------------------------------------------
// Problem constants
// ---------------------------------------------------------------------------
constexpr int Bn = 8;
constexpr int Sn = 2048;
constexpr int Dn = 512;
constexpr float ATTN_SCALE = 0.125f;   // 1/sqrt(64), per reference

typedef __attribute__((ext_vector_type(16))) __bf16 v16bf;
typedef __attribute__((ext_vector_type(8)))  float  v8f;

union FragBF { v16bf v; uint4 q[2]; };

#if defined(__has_builtin)
#if __has_builtin(__builtin_amdgcn_global_load_async_to_lds_b128)
#define HAVE_ASYNC_LDS 1
#endif
#endif

#ifdef HAVE_ASYNC_LDS
// Probe-derived prototype: (global v4i*, local v4i*, imm int, imm int)
typedef int v4i __attribute__((vector_size(16)));
typedef __attribute__((address_space(1))) v4i as1_v4i;
typedef __attribute__((address_space(3))) v4i as3_v4i;
#endif

__device__ __forceinline__ unsigned short f2bf(float x) {
  unsigned int u = __float_as_uint(x);
  return (unsigned short)((u + 0x7fffu + ((u >> 16) & 1u)) >> 16);  // RNE
}

// Load a 16x32 bf16 WMMA fragment (A layout; also valid for B when the
// matrix is stored "lane-dim-major, contiguous along K", i.e. N x K row-major).
// Per ISA 7.12.2: lanes 0-15 hold K {k0..k0+7, k0+16..k0+23},
//                 lanes 16-31 hold K {k0+8..k0+15, k0+24..k0+31}.
__device__ __forceinline__ v16bf load_frag(const unsigned short* base, int row0,
                                           int ld, int k0, int lane) {
  const int m = lane & 15;
  const int h = lane >> 4;
  const unsigned short* p = base + (size_t)(row0 + m) * ld + k0 + 8 * h;
  FragBF f;
  f.q[0] = *(const uint4*)(p);        // K chunk [k0+8h, +8)
  f.q[1] = *(const uint4*)(p + 16);   // K chunk [k0+16+8h, +8)
  return f.v;
}

__device__ __forceinline__ v8f wmma_bf16(v16bf a, v16bf b, v8f c) {
  return __builtin_amdgcn_wmma_f32_16x16x32_bf16(
      /*neg_a=*/false, a, /*neg_b=*/false, b,
      /*c_mod=*/(short)0, c, /*reuse_a=*/false, /*reuse_b=*/false);
}

// ---------------------------------------------------------------------------
// Kernel 0a: x f32 -> bf16 (row-major [B*S, D])
// ---------------------------------------------------------------------------
__global__ void cvt_x(const float4* __restrict__ x, unsigned short* __restrict__ xb,
                      int n4) {
  const int i = blockIdx.x * blockDim.x + threadIdx.x;
  if (i >= n4) return;
  float4 v = x[i];
  union { unsigned short u[4]; uint2 d; } o;
  o.u[0] = f2bf(v.x); o.u[1] = f2bf(v.y);
  o.u[2] = f2bf(v.z); o.u[3] = f2bf(v.w);
  *(uint2*)(xb + (size_t)i * 4) = o.d;
}

// ---------------------------------------------------------------------------
// Kernel 0b: kernel f32 [3][D][E] -> bf16 transposed Wt [3][E][D]
// ---------------------------------------------------------------------------
__global__ void cvt_w(const float* __restrict__ k, unsigned short* __restrict__ wt) {
  const int w   = blockIdx.z;
  const int idx = blockIdx.x * blockDim.x + threadIdx.x;  // 0 .. D*D-1
  const int d = idx >> 9;
  const int e = idx & 511;
  wt[(size_t)w * Dn * Dn + (size_t)e * Dn + d] =
      f2bf(k[(size_t)w * Dn * Dn + (size_t)d * Dn + e]);
}

// ---------------------------------------------------------------------------
// Kernel 1: QKV projection. One wave computes a 16x64 output strip (4 tiles):
// per K-step the A fragment is loaded ONCE and reused across 4 WMMAs.
//   Q -> Qb [B*S, D] bf16, pre-scaled by 1/sqrt(64)
//   K -> Kb [B*S, D] bf16
//   V -> Vt [B][D][S] bf16 (transposed for the PV B-fragment)
// ---------------------------------------------------------------------------
__global__ __launch_bounds__(256, 1)
void qkv_proj(const unsigned short* __restrict__ Xb,
              const unsigned short* __restrict__ Wt,
              unsigned short* __restrict__ Qb,
              unsigned short* __restrict__ Kb,
              unsigned short* __restrict__ Vt) {
  const int lane = threadIdx.x & 31;
  const int wave = threadIdx.x >> 5;
  const int row0 = blockIdx.x * 16;        // row tile in flattened B*S
  const int n0   = wave * 64;              // 64-column strip per wave
  const int w    = blockIdx.z;             // 0=Q,1=K,2=V
  const unsigned short* Wb = Wt + (size_t)w * Dn * Dn;

  const v8f zero = {0.f, 0.f, 0.f, 0.f, 0.f, 0.f, 0.f, 0.f};
  v8f acc[4];
  #pragma unroll
  for (int t = 0; t < 4; ++t) acc[t] = zero;

  #pragma unroll 2
  for (int d0 = 0; d0 < Dn; d0 += 32) {
    const v16bf a = load_frag(Xb, row0, Dn, d0, lane);   // reused 4x
    #pragma unroll
    for (int t = 0; t < 4; ++t) {
      v16bf bm = load_frag(Wb, n0 + t * 16, Dn, d0, lane);
      acc[t] = wmma_bf16(a, bm, acc[t]);
    }
  }

  // C/D layout: VGPR r -> row m = r + 8*(lane>=16); column = lane&15
  const int m_off = (lane & 16) ? 8 : 0;
  #pragma unroll
  for (int t = 0; t < 4; ++t) {
    const int col = n0 + t * 16 + (lane & 15);
    if (w == 0) {
      #pragma unroll
      for (int r = 0; r < 8; ++r)
        Qb[(size_t)(row0 + r + m_off) * Dn + col] = f2bf(acc[t][r] * ATTN_SCALE);
    } else if (w == 1) {
      #pragma unroll
      for (int r = 0; r < 8; ++r)
        Kb[(size_t)(row0 + r + m_off) * Dn + col] = f2bf(acc[t][r]);
    } else {
      #pragma unroll
      for (int r = 0; r < 8; ++r) {
        const int row = row0 + r + m_off;
        const int bb  = row >> 11;          // / Sn
        const int s   = row & (Sn - 1);
        Vt[((size_t)bb * Dn + col) * Sn + s] = f2bf(acc[t][r]);
      }
    }
  }
}

// ---------------------------------------------------------------------------
// Kernel 2: flash attention.
// Block = 256 threads = 8 waves: 2 q-rowgroups (16 rows each) x 4 column
// groups (128 out cols each). The 32x512 bf16 Q block is staged into LDS once
// via GLOBAL_LOAD_ASYNC_TO_LDS_B128 (ASYNCcnt) and re-read from LDS by every
// key iteration. Keys stream in blocks of 64 with online softmax;
// probabilities are transposed through LDS into WMMA A-fragment layout.
// ---------------------------------------------------------------------------
__global__ __launch_bounds__(256, 1)
void flash_attn(const unsigned short* __restrict__ Qb,
                const unsigned short* __restrict__ Kb,
                const unsigned short* __restrict__ Vt,
                float* __restrict__ out) {
  __shared__ __align__(16) unsigned short Qs[2][16][Dn];  // 32 KB, loop-invariant
  __shared__ __align__(16) float          Sbuf[2][16][64]; // scores (f32)
  __shared__ __align__(16) unsigned short Pbuf[2][16][64]; // probs (bf16)
  __shared__ float Abuf[2][16];                            // per-row alpha
  __shared__ float Lbuf[2][16];                            // per-row 1/l

  const int tid  = threadIdx.x;
  const int lane = tid & 31;
  const int wave = tid >> 5;
  const int qt = wave >> 2;          // 0..1 rowgroup
  const int ct = wave & 3;           // 0..3 column group
  const int b  = blockIdx.y;
  const int row0 = blockIdx.x * 32 + qt * 16;

  // ---- stage Q block (32 rows x 512) into LDS, once ----
  {
    unsigned short* lbase = &Qs[0][0][0];
    const unsigned short* gbase = Qb + (size_t)(b * Sn + blockIdx.x * 32) * Dn;
#ifdef HAVE_ASYNC_LDS
    #pragma unroll
    for (int s = 0; s < 8; ++s) {
      const int off = (s * 256 + tid) * 8;   // 16 bytes per thread per sweep
      __builtin_amdgcn_global_load_async_to_lds_b128(
          (as1_v4i*)(gbase + off), (as3_v4i*)(lbase + off), 0, 0);
    }
#if __has_builtin(__builtin_amdgcn_s_wait_asynccnt)
    __builtin_amdgcn_s_wait_asynccnt(0);
#else
    asm volatile("s_wait_asynccnt 0x0" ::: "memory");
#endif
#else
    #pragma unroll
    for (int s = 0; s < 8; ++s) {
      const int off = (s * 256 + tid) * 8;
      *(uint4*)(lbase + off) = *(const uint4*)(gbase + off);
    }
#endif
    __syncthreads();
  }

  const int r     = lane & 15;       // softmax lane -> row
  const int ch    = lane >> 4;       // softmax lane -> column half (32 cols)
  const int m_off = (lane & 16) ? 8 : 0;

  const v8f zero = {0.f, 0.f, 0.f, 0.f, 0.f, 0.f, 0.f, 0.f};
  v8f acc[8];
  #pragma unroll
  for (int t = 0; t < 8; ++t) acc[t] = zero;

  float m_row = -3.0e38f;
  float l_row = 0.f;

  for (int kb = 0; kb < Sn; kb += 64) {
    // ---- scores: 16 q-rows x 16 keys for this wave (keys kb + ct*16 ..) ----
    v8f sacc = zero;
    #pragma unroll 4
    for (int d0 = 0; d0 < Dn; d0 += 32) {
      v16bf qa  = load_frag(&Qs[qt][0][0], 0, Dn, d0, lane);      // LDS
      v16bf kbf = load_frag(Kb, b * Sn + kb + ct * 16, Dn, d0, lane);
      sacc = wmma_bf16(qa, kbf, sacc);
    }
    if (kb + 64 < Sn)  // hint the next K block into cache
      __builtin_prefetch(Kb + (size_t)(b * Sn + kb + 64 + ct * 16 + r) * Dn, 0, 0);

    __syncthreads();  // (A) previous iteration's Sbuf/Pbuf consumers done
    {
      const int col = ct * 16 + (lane & 15);
      #pragma unroll
      for (int rr = 0; rr < 8; ++rr)
        Sbuf[qt][rr + m_off][col] = sacc[rr];
    }
    __syncthreads();  // (B) Sbuf ready

    // ---- online softmax over this 16x64 block (redundant per rowgroup) ----
    float4 sv[8];
    const float* srow = &Sbuf[qt][r][ch * 32];
    #pragma unroll
    for (int j = 0; j < 8; ++j) sv[j] = ((const float4*)srow)[j];

    float tmax = -3.0e38f;
    #pragma unroll
    for (int j = 0; j < 8; ++j)
      tmax = fmaxf(tmax, fmaxf(fmaxf(sv[j].x, sv[j].y), fmaxf(sv[j].z, sv[j].w)));
    tmax = fmaxf(tmax, __shfl_xor(tmax, 16, 32));

    const float m_new = fmaxf(m_row, tmax);
    const float alpha = __expf(m_row - m_new);
    float psum = 0.f;
    unsigned short pl[32];
    #pragma unroll
    for (int j = 0; j < 8; ++j) {
      const float p0 = __expf(sv[j].x - m_new);
      const float p1 = __expf(sv[j].y - m_new);
      const float p2 = __expf(sv[j].z - m_new);
      const float p3 = __expf(sv[j].w - m_new);
      psum += (p0 + p1) + (p2 + p3);
      pl[4*j+0] = f2bf(p0); pl[4*j+1] = f2bf(p1);
      pl[4*j+2] = f2bf(p2); pl[4*j+3] = f2bf(p3);
    }
    psum += __shfl_xor(psum, 16, 32);
    l_row = l_row * alpha + psum;
    m_row = m_new;

    if (ct == 0) {
      uint4* pd = (uint4*)&Pbuf[qt][r][ch * 32];
      const uint4* ps = (const uint4*)pl;
      #pragma unroll
      for (int j = 0; j < 4; ++j) pd[j] = ps[j];
      if (lane < 16) Abuf[qt][lane] = alpha;   // lane == row here
    }
    __syncthreads();  // (C) Pbuf/Abuf ready

    // ---- rescale accumulator by alpha (broadcast via LDS) ----
    #pragma unroll
    for (int rr = 0; rr < 8; ++rr) {
      const float al = Abuf[qt][rr + m_off];
      #pragma unroll
      for (int t = 0; t < 8; ++t) acc[t][rr] *= al;
    }

    // ---- PV: acc += P[16x64] @ V[64 x 128-cols]  (16 WMMAs/wave) ----
    #pragma unroll
    for (int kk = 0; kk < 64; kk += 32) {
      v16bf pa = load_frag(&Pbuf[qt][0][0], 0, 64, kk, lane);     // LDS, reused 8x
      #pragma unroll
      for (int t = 0; t < 8; ++t) {
        v16bf vb = load_frag(Vt + (size_t)b * Dn * Sn,
                             ct * 128 + t * 16, Sn, kb + kk, lane);
        acc[t] = wmma_bf16(pa, vb, acc[t]);
      }
    }
  }

  __syncthreads();
  if (ct == 0 && lane < 16) Lbuf[qt][lane] = 1.0f / l_row;
  __syncthreads();

  const int qrow = b * Sn + row0;
  const int col0 = ct * 128 + (lane & 15);
  #pragma unroll
  for (int rr = 0; rr < 8; ++rr) {
    const float inv = Lbuf[qt][rr + m_off];
    float* orow = out + (size_t)(qrow + rr + m_off) * Dn;
    #pragma unroll
    for (int t = 0; t < 8; ++t)
      orow[col0 + t * 16] = acc[t][rr] * inv;
  }
}

// ---------------------------------------------------------------------------
// Launch
// ---------------------------------------------------------------------------
extern "C" void kernel_launch(void* const* d_in, const int* in_sizes, int n_in,
                              void* d_out, int out_size, void* d_ws, size_t ws_size,
                              hipStream_t stream) {
  (void)in_sizes; (void)n_in; (void)out_size; (void)ws_size;
  const float* x    = (const float*)d_in[0];   // [B, S, D] f32
  const float* kern = (const float*)d_in[1];   // [3, D, D] f32
  float* out = (float*)d_out;                  // [B, S, D] f32

  char* ws = (char*)d_ws;
  constexpr size_t XB = (size_t)Bn * Sn * Dn * sizeof(unsigned short); // 16 MiB
  constexpr size_t WT = (size_t)3 * Dn * Dn * sizeof(unsigned short);  // 1.5 MiB
  unsigned short* xb = (unsigned short*)(ws);
  unsigned short* wt = (unsigned short*)(ws + XB);
  unsigned short* qb = (unsigned short*)(ws + XB + WT);
  unsigned short* kb = (unsigned short*)(ws + XB + WT + XB);
  unsigned short* vt = (unsigned short*)(ws + XB + WT + 2 * XB);
  // total workspace: 4*16MiB + 1.5MiB ~= 65.5 MiB

  const int n4 = Bn * Sn * Dn / 4;
  cvt_x<<<dim3((n4 + 255) / 256), 256, 0, stream>>>((const float4*)x, xb, n4);
  cvt_w<<<dim3(Dn * Dn / 256, 1, 3), 256, 0, stream>>>(kern, wt);
  qkv_proj<<<dim3(Bn * Sn / 16, 1, 3), 256, 0, stream>>>(xb, wt, qb, kb, vt);
  flash_attn<<<dim3(Sn / 32, Bn), 256, 0, stream>>>(qb, kb, vt, out);
}
